// LSTMCell_56642028699866
// MI455X (gfx1250) — matrix operand order repaired and checked by reading, not compiled, not run on hardware
//
#include <hip/hip_runtime.h>
#include <hip/hip_bf16.h>

// LSTM cell fused kernel for MI455X (gfx1250), bf16 WMMA with f32 accumulate.
// Workspace layout (requires >= 32 MB):
//   [0 .. 8M)  u16 : hx_bf16   = concat(x,h)  [4096][2048]
//   [8M..16M)  u16 : wt_bf16   = W^T per gate [4][1024][2048]  (N-major)

typedef unsigned short u16;
typedef __attribute__((ext_vector_type(16))) __bf16 v16bf;
typedef __attribute__((ext_vector_type(8)))  float  v8f;
typedef int v4i_vs __attribute__((vector_size(16)));   // matches builtin param

#define BATCH 4096
#define KDIM  2048
#define HDIM  1024
#define LROW  40   // padded LDS row (u16) for a 32-element K row: 80B stride

#if defined(__has_builtin)
#  if __has_builtin(__builtin_amdgcn_global_load_async_to_lds_b128)
#    define HAVE_ASYNC_LDS 1
#  endif
#endif
#ifndef HAVE_ASYNC_LDS
#  define HAVE_ASYNC_LDS 0
#endif

__device__ __forceinline__ u16 f32_to_bf16(float f) {
  unsigned int u = __float_as_uint(f);
  u += 0x7FFFu + ((u >> 16) & 1u);   // round-to-nearest-even
  return (u16)(u >> 16);
}

__device__ __forceinline__ float sigmoid_f(float x) {
  return 1.0f / (1.0f + __expf(-x));
}
__device__ __forceinline__ float tanh_f(float x) {
  return 1.0f - 2.0f / (__expf(2.0f * x) + 1.0f);   // stable at +/-inf
}

// Fence-free workgroup barrier: our pipeline's cross-wave ordering is fully
// handled by s_wait_asynccnt (producers) and dscnt waits (consumers), so we
// avoid __syncthreads()'s implied s_wait_loadcnt 0 which would serialize the
// wave-private A-prefetch loads with the barrier.
__device__ __forceinline__ void wg_barrier() {
  asm volatile("" ::: "memory");
#if defined(__has_builtin) && __has_builtin(__builtin_amdgcn_s_barrier)
  __builtin_amdgcn_s_barrier();
#else
  asm volatile("s_barrier_signal -1\n\ts_barrier_wait -1" ::: "memory");
#endif
  asm volatile("" ::: "memory");
}

// ---------------------------------------------------------------------------
// Pass 1: hx_bf16[b][k] = bf16( k<1024 ? x[b][k] : h[b][k-1024] )
// ---------------------------------------------------------------------------
__global__ __launch_bounds__(256) void build_hx_kernel(
    const float* __restrict__ x, const float* __restrict__ h,
    u16* __restrict__ hx) {
  int t   = blockIdx.x * blockDim.x + threadIdx.x;
  int row = t >> 8;
  int kc  = (t & 255) << 3;
  const float* src = (kc < HDIM) ? (x + row * HDIM + kc)
                                 : (h + row * HDIM + (kc - HDIM));
  float4 f0 = ((const float4*)src)[0];
  float4 f1 = ((const float4*)src)[1];
  union { u16 s[8]; uint4 q; } o;
  o.s[0] = f32_to_bf16(f0.x); o.s[1] = f32_to_bf16(f0.y);
  o.s[2] = f32_to_bf16(f0.z); o.s[3] = f32_to_bf16(f0.w);
  o.s[4] = f32_to_bf16(f1.x); o.s[5] = f32_to_bf16(f1.y);
  o.s[6] = f32_to_bf16(f1.z); o.s[7] = f32_to_bf16(f1.w);
  *(uint4*)(hx + row * KDIM + kc) = o.q;
}

// ---------------------------------------------------------------------------
// Pass 2: transpose + convert weights: wt[g][n][k] = bf16(W_g[k][n])
// ---------------------------------------------------------------------------
__global__ __launch_bounds__(256) void convert_w_kernel(
    const float* __restrict__ w0, const float* __restrict__ w1,
    const float* __restrict__ w2, const float* __restrict__ w3,
    u16* __restrict__ wt) {
  __shared__ u16 tile[64][72];
  const float* Wg[4] = {w0, w1, w2, w3};
  const float* W = Wg[blockIdx.z];
  int k0 = blockIdx.x * 64, n0 = blockIdx.y * 64;
  int tx = threadIdx.x & 63, ty = threadIdx.x >> 6;
#pragma unroll
  for (int i = 0; i < 16; ++i) {
    int kl = ty * 16 + i;
    tile[kl][tx] = f32_to_bf16(W[(k0 + kl) * HDIM + n0 + tx]);
  }
  __syncthreads();
  int gbase = blockIdx.z * HDIM;
#pragma unroll
  for (int i = 0; i < 16; ++i) {
    int nl = ty * 16 + i;
    wt[(gbase + n0 + nl) * KDIM + k0 + tx] = tile[tx][nl];
  }
}

// ---------------------------------------------------------------------------
// Pass 3: fused 4-gate GEMM (bf16 WMMA, f32 acc) + LSTM elementwise.
// Block 256 thr = 8 waves (4 M x 2 N).  Block tile 128(M) x 64(N).
// B weight tiles (4 gates x 64 cols x 32 K) are staged in LDS, double
// buffered with gfx1250 async global->LDS copies, shared by all 4 M-waves.
// A fragments are register double-buffered from global (L2-resident).
// ---------------------------------------------------------------------------
union Frag { v16bf v; uint4 u[2]; };

__global__ __launch_bounds__(256) void lstm_gemm_wmma_kernel(
    const u16* __restrict__ hx, const u16* __restrict__ wt,
    const float* __restrict__ cin,
    const float* __restrict__ bF, const float* __restrict__ bI,
    const float* __restrict__ bS, const float* __restrict__ bP,
    float* __restrict__ hout) {
  __shared__ __align__(16) u16 ldsB[2][4 * 64 * LROW];   // 2 x 20 KB

  const int tid  = threadIdx.x;
  const int lane = tid & 31;
  const int wave = tid >> 5;
  const int wr   = wave >> 1;            // 0..3  (M)
  const int wc   = wave & 1;             // 0..1  (N)
  const int half = lane >> 4;
  const int ln   = lane & 15;
  const int mBase = blockIdx.y * 128 + wr * 32;
  const int nBlk  = blockIdx.x * 64;

  // Staging role: thread (sg,sn) copies one (gate,col) K-row of 64B per step.
  const int sg = tid >> 6;               // gate 0..3
  const int sn = tid & 63;               // col  0..63
  const u16* sGlb = wt + ((size_t)sg * HDIM + nBlk + sn) * KDIM;
  u16* sLds0 = &ldsB[0][(sg * 64 + sn) * LROW];
  u16* sLds1 = &ldsB[1][(sg * 64 + sn) * LROW];

  // A fragment pointers (ISA 7.12.2 16x32 bf16 layout).
  const u16* aPtr[2];
#pragma unroll
  for (int mi = 0; mi < 2; ++mi)
    aPtr[mi] = hx + (mBase + mi * 16 + ln) * KDIM + half * 8;

  // Per-lane constant part of the B fragment LDS address.
  const u16* bBase = &ldsB[0][(wc * 32 + ln) * LROW + half * 8];

  v8f acc[4][2][2] = {};                 // [gate][mi][ni]
  Frag aF[2][2];                         // [buf][mi]

#if HAVE_ASYNC_LDS
#define STAGE(BUFPTR, K0)                                                    \
  do {                                                                       \
    _Pragma("unroll") for (int j = 0; j < 4; ++j) {                          \
      __builtin_amdgcn_global_load_async_to_lds_b128(                        \
          (v4i_vs*)(sGlb + (K0) + j * 8),                                    \
          (v4i_vs*)((BUFPTR) + j * 8), 0, 0);                                \
    }                                                                        \
  } while (0)
#define WAIT_STAGE() asm volatile("s_wait_asynccnt 0x0" ::: "memory")
#else
#define STAGE(BUFPTR, K0)                                                    \
  do {                                                                       \
    uint4 t0 = *(const uint4*)(sGlb + (K0));                                 \
    uint4 t1 = *(const uint4*)(sGlb + (K0) + 8);                             \
    uint4 t2 = *(const uint4*)(sGlb + (K0) + 16);                            \
    uint4 t3 = *(const uint4*)(sGlb + (K0) + 24);                            \
    *(uint4*)((BUFPTR))      = t0;                                           \
    *(uint4*)((BUFPTR) + 8)  = t1;                                           \
    *(uint4*)((BUFPTR) + 16) = t2;                                           \
    *(uint4*)((BUFPTR) + 24) = t3;                                           \
    asm volatile("s_wait_dscnt 0x0" ::: "memory");                           \
  } while (0)
#define WAIT_STAGE() do {} while (0)
#endif

#define LOAD_A(BUF, K0)                                                      \
  do {                                                                       \
    _Pragma("unroll") for (int mi = 0; mi < 2; ++mi) {                       \
      aF[BUF][mi].u[0] = *(const uint4*)(aPtr[mi] + (K0));                   \
      aF[BUF][mi].u[1] = *(const uint4*)(aPtr[mi] + (K0) + 16);              \
    }                                                                        \
  } while (0)

#define CONSUME(BUF)                                                         \
  do {                                                                       \
    _Pragma("unroll") for (int g = 0; g < 4; ++g) {                          \
      _Pragma("unroll") for (int ni = 0; ni < 2; ++ni) {                     \
        const u16* lp =                                                      \
            bBase + (BUF) * (4 * 64 * LROW) + (g * 64 + ni * 16) * LROW;     \
        Frag b;                                                              \
        b.u[0] = *(const uint4*)lp;                                          \
        b.u[1] = *(const uint4*)(lp + 16);                                   \
        _Pragma("unroll") for (int mi = 0; mi < 2; ++mi) {                   \
          acc[g][mi][ni] = __builtin_amdgcn_wmma_f32_16x16x32_bf16(          \
              false, aF[BUF][mi].v, false, b.v, (short)0, acc[g][mi][ni],    \
              false, false);                                                 \
        }                                                                    \
      }                                                                      \
    }                                                                        \
  } while (0)

// One K-step: A loads for next step fly first (and stay in flight across the
// fence-free barrier), then s_wait_asynccnt + barrier publishes the current
// LDS buffer, next-buffer async copies are issued, and the 16 WMMAs consume
// the current buffer while those copies are in flight.
#define PIPE(CUR, NXT, SLDSNXT, K0)                                          \
  do {                                                                       \
    const int kn = (K0) + 32;                                                \
    if (kn < KDIM) LOAD_A(NXT, kn);                                          \
    WAIT_STAGE();                                                            \
    wg_barrier();                                                            \
    if (kn < KDIM) STAGE(SLDSNXT, kn);                                       \
    CONSUME(CUR);                                                            \
  } while (0)

  LOAD_A(0, 0);
  STAGE(sLds0, 0);
  for (int it = 0; it < KDIM / 64; ++it) {
    const int k0 = it * 64;
    PIPE(0, 1, sLds1, k0);
    PIPE(1, 0, sLds0, k0 + 32);
  }
#undef PIPE
#undef CONSUME
#undef LOAD_A
#undef STAGE
#undef WAIT_STAGE

  // Epilogue: bias + gates + cell update.  C/D layout: VGPR r <-> M=r+8*half.
#pragma unroll
  for (int mi = 0; mi < 2; ++mi) {
#pragma unroll
    for (int ni = 0; ni < 2; ++ni) {
      const int n  = nBlk + wc * 32 + ni * 16 + ln;
      const float vbf = bF[n], vbi = bI[n], vbs = bS[n], vbp = bP[n];
#pragma unroll
      for (int r = 0; r < 8; ++r) {
        const int m = mBase + mi * 16 + half * 8 + r;
        const float fg = sigmoid_f(acc[0][mi][ni][r] + vbf);
        const float ig = sigmoid_f(acc[1][mi][ni][r] + vbi);
        const float sg2 = sigmoid_f(acc[2][mi][ni][r] + vbs);
        const float pg = tanh_f   (acc[3][mi][ni][r] + vbp);
        const float cn = cin[m * HDIM + n] * fg + ig * pg;
        hout[m * HDIM + n] = tanh_f(cn) * sg2;
      }
    }
  }
}

// ---------------------------------------------------------------------------
extern "C" void kernel_launch(void* const* d_in, const int* in_sizes, int n_in,
                              void* d_out, int out_size, void* d_ws,
                              size_t ws_size, hipStream_t stream) {
  const float* x  = (const float*)d_in[0];
  const float* h  = (const float*)d_in[1];
  const float* c  = (const float*)d_in[2];
  const float* Wf = (const float*)d_in[3];
  const float* bF = (const float*)d_in[4];
  const float* Wi = (const float*)d_in[5];
  const float* bI = (const float*)d_in[6];
  const float* Ws = (const float*)d_in[7];
  const float* bS = (const float*)d_in[8];
  const float* Wp = (const float*)d_in[9];
  const float* bP = (const float*)d_in[10];
  float* out = (float*)d_out;

  u16* hx_bf = (u16*)d_ws;                                // 16 MB
  u16* wt_bf = hx_bf + (size_t)BATCH * KDIM;              // 16 MB

  build_hx_kernel<<<(BATCH * KDIM / 8) / 256, 256, 0, stream>>>(x, h, hx_bf);

  dim3 gw(KDIM / 64, HDIM / 64, 4);
  convert_w_kernel<<<gw, 256, 0, stream>>>(Wf, Wi, Ws, Wp, wt_bf);

  dim3 gg(HDIM / 64, BATCH / 128);
  lstm_gemm_wmma_kernel<<<gg, 256, 0, stream>>>(hx_bf, wt_bf, c, bF, bI, bS,
                                                bP, out);
}
